// SASAEncoder_30915174597288
// MI455X (gfx1250) — matrix operand order repaired and checked by reading, not compiled, not run on hardware
//
#include <hip/hip_runtime.h>

typedef __attribute__((ext_vector_type(16))) _Float16 v16h;
typedef __attribute__((ext_vector_type(8)))  _Float16 h8;
typedef __attribute__((ext_vector_type(8)))  float    v8f;
typedef __attribute__((ext_vector_type(4)))  unsigned int u32x4;
typedef __attribute__((ext_vector_type(8)))  unsigned int u32x8;

#define NPIX 16384   // 4*64*64
#define CCH  128
#define DKH  16
#define KWIN 7
#define HDIM 64
#define WDIM 64

union H16x16 { v16h v; h8 h[2]; };
union F32x8  { v8f  v; float e[8]; };

// ---------------------------------------------------------------------------
// Tensor Data Mover: issue a 2D f16 tile load Global->LDS (ISA 8.3/8.4 D#).
// Groups 2/3 omitted (2D tensor => VADDR2/VADDR3 = NULL form).
// ---------------------------------------------------------------------------
__device__ __forceinline__ void tdm_load_2d_f16(unsigned lds_addr,
                                                const _Float16* gptr,
                                                unsigned tensor_d0, unsigned tensor_d1,
                                                unsigned tile_d0, unsigned tile_d1,
                                                unsigned stride0) {
    const unsigned long long ga = (unsigned long long)(uintptr_t)gptr;
    u32x4 g0;
    g0[0] = 1u;                                               // count=1, user desc
    g0[1] = lds_addr;                                         // LDS byte address
    g0[2] = (unsigned)ga;                                     // global_addr[31:0]
    g0[3] = ((unsigned)(ga >> 32) & 0x01FFFFFFu) | (2u << 30);// addr[56:32] | type=2
    u32x8 g1;
    g1[0] = (1u << 16);                                       // data_size=1 (2 bytes)
    g1[1] = (tensor_d0 & 0xFFFFu) << 16;                      // tensor_dim0[15:0]
    g1[2] = (tensor_d0 >> 16) | ((tensor_d1 & 0xFFFFu) << 16);
    g1[3] = (tensor_d1 >> 16) | ((tile_d0 & 0xFFFFu) << 16);  // tile_dim0
    g1[4] = (tile_d1 & 0xFFFFu);                              // tile_dim1 (tile_dim2=0)
    g1[5] = stride0;                                          // tensor_dim0_stride[31:0]
    g1[6] = 0u;
    g1[7] = 0u;
    asm volatile("tensor_load_to_lds %0, %1" :: "s"(g0), "s"(g1) : "memory");
}

// ---------------------------------------------------------------------------
// Prep 0a: convert+transpose the four 128x128 weights into f16 Wt[w][out][in]
// ---------------------------------------------------------------------------
__global__ void prep_weights(const float* __restrict__ Wq, const float* __restrict__ Wk,
                             const float* __restrict__ Wv, const float* __restrict__ Wo,
                             _Float16* __restrict__ Wt) {
    const int idx = blockIdx.x * 256 + threadIdx.x;   // 0..65535
    const int w = idx >> 14;
    const int rc = idx & 16383;
    const int o = rc >> 7, i = rc & 127;
    const float* W = (w == 0) ? Wq : (w == 1) ? Wk : (w == 2) ? Wv : Wo;
    Wt[idx] = (_Float16)W[i * CCH + o];
}

// Prep 0b: x f32 -> f16
__global__ void convert_x(const float* __restrict__ x, _Float16* __restrict__ x16) {
    const int idx = blockIdx.x * 256 + threadIdx.x;
    x16[idx] = (_Float16)x[idx];
}

// ---------------------------------------------------------------------------
// Kernel 1: fused QKV projection. Block=256 (8 waves), grid=(1024 M-tiles, 3).
// Wave 0 TDM-loads the 16x128 f16 A-tile into LDS; wave w computes N-tile w
// of {Q,K,V}[blockIdx.y] with 4x v_wmma_f32_16x16x32_f16.
// ---------------------------------------------------------------------------
__global__ void qkv_proj_wmma(const _Float16* __restrict__ x16,
                              const _Float16* __restrict__ Wt,
                              float* __restrict__ qkv) {
    __shared__ __align__(16) _Float16 xs[16 * CCH];
    const int t = threadIdx.x;
    const int lane = t & 31;
    const int half = lane >> 4;
    const int hl   = lane & 15;
    const int wave = t >> 5;                  // 0..7 -> N tile
    const int row0 = blockIdx.x * 16;
    const int wsel = blockIdx.y;              // 0=Q 1=K 2=V
    const _Float16* Wm = Wt + (size_t)wsel * CCH * CCH;
    const int ncol0 = wave * 16;

    if (t < 32) {
        tdm_load_2d_f16((unsigned)(uintptr_t)xs, x16 + (size_t)row0 * CCH,
                        CCH, NPIX, CCH, 16, CCH);
        __builtin_amdgcn_s_wait_tensorcnt(0);
    }
    __syncthreads();

    F32x8 acc; acc.v = (v8f){0.f,0.f,0.f,0.f,0.f,0.f,0.f,0.f};
    #pragma unroll
    for (int kk = 0; kk < 4; ++kk) {          // K = 128 in steps of 32
        const int kbase = kk * 32;
        H16x16 a, b;
        // A 16x32 f16: lane half<16 holds K {0..7,16..23}, half>=16 {8..15,24..31}
        a.h[0] = *(const h8*)(xs + hl * CCH + kbase + half * 8);
        a.h[1] = *(const h8*)(xs + hl * CCH + kbase + 16 + half * 8);
        // B 32x16 f16: lane = column; lanes 0-15 hold K {0..7,16..23}, 16-31 hold K {8..15,24..31}
        const _Float16* bp = Wm + (ncol0 + hl) * CCH + kbase + half * 8;
        b.h[0] = *(const h8*)(bp);
        b.h[1] = *(const h8*)(bp + 16);
        acc.v = __builtin_amdgcn_wmma_f32_16x16x32_f16(false, a.v, false, b.v,
                                                       (short)0, acc.v, false, false);
    }
    // D layout: VGPR r -> row 8*half + r, col = hl
    float* dst = qkv + (size_t)wsel * NPIX * CCH;
    #pragma unroll
    for (int r = 0; r < 8; ++r)
        dst[(row0 + half * 8 + r) * CCH + ncol0 + hl] = acc.e[r];
}

// ---------------------------------------------------------------------------
// Kernel 2: 7x7 sliding-window attention with relative-position logits.
// grid = 16384 pixels, block = 128 = (head n = t>>4, dim d = t&15).
// Online softmax; 16-lane shfl_xor reduction for per-head dot products.
// Zero-padded neighbors: k=v=0 but the q.rel logit still participates.
// ---------------------------------------------------------------------------
__global__ void sasa_attn(const float* __restrict__ q,
                          const float* __restrict__ kmap,
                          const float* __restrict__ vmap,
                          const float* __restrict__ rel_h,
                          const float* __restrict__ rel_w,
                          _Float16* __restrict__ att16) {
    const int p = blockIdx.x;
    const int t = threadIdx.x;
    const int n = t >> 4;
    const int d = t & 15;
    const int b = p >> 12;
    const int h = (p >> 6) & 63;
    const int w = p & 63;

    const float qd = q[p * CCH + n * DKH + d];
    float m = -3.0e38f, s = 0.f, acc = 0.f;

    for (int kh = 0; kh < KWIN; ++kh) {
        const int h2 = h + kh - 3;
        const bool hok = (h2 >= 0) && (h2 < HDIM);
        for (int kw = 0; kw < KWIN; ++kw) {
            const int w2 = w + kw - 3;
            const bool inb = hok && (w2 >= 0) && (w2 < WDIM);
            float kv = 0.f, vv = 0.f;
            if (inb) {
                const int np_ = ((b << 6) + h2) * 64 + w2;
                kv = kmap[np_ * CCH + n * DKH + d];
                vv = vmap[np_ * CCH + n * DKH + d];
            }
            const float rv = (d < 8) ? rel_h[kh * 8 + d] : rel_w[kw * 8 + (d - 8)];
            float part = qd * (kv + rv);
            #pragma unroll
            for (int off = 1; off < 16; off <<= 1)
                part += __shfl_xor(part, off, 32);
            const float lg = part;
            const float mN = fmaxf(m, lg);
            const float sc = __expf(m - mN);
            const float e  = __expf(lg - mN);
            acc = acc * sc + e * vv;
            s   = s * sc + e;
            m   = mN;
        }
    }
    att16[p * CCH + n * DKH + d] = (_Float16)(acc / s);
}

// ---------------------------------------------------------------------------
// Kernel 3: y = att * Wo (WMMA, A-tile via TDM) fused with LayerNorm.
// grid = 1024 M-tiles, block = 256 (8 waves; wave w -> column tile w).
// ---------------------------------------------------------------------------
__global__ void out_proj_ln_wmma(const _Float16* __restrict__ att16,
                                 const _Float16* __restrict__ WoT,
                                 const float* __restrict__ gamma,
                                 const float* __restrict__ beta,
                                 float* __restrict__ out) {
    __shared__ __align__(16) _Float16 as16[16 * CCH];
    __shared__ float ytile[16 * 132];
    const int t    = threadIdx.x;
    const int wave = t >> 5;
    const int lane = t & 31;
    const int half = lane >> 4;
    const int hl   = lane & 15;
    const int row0  = blockIdx.x * 16;
    const int ncol0 = wave * 16;

    if (t < 32) {
        tdm_load_2d_f16((unsigned)(uintptr_t)as16, att16 + (size_t)row0 * CCH,
                        CCH, NPIX, CCH, 16, CCH);
        __builtin_amdgcn_s_wait_tensorcnt(0);
    }
    __syncthreads();

    F32x8 acc; acc.v = (v8f){0.f,0.f,0.f,0.f,0.f,0.f,0.f,0.f};
    #pragma unroll
    for (int kk = 0; kk < 4; ++kk) {
        const int kbase = kk * 32;
        H16x16 a, b;
        a.h[0] = *(const h8*)(as16 + hl * CCH + kbase + half * 8);
        a.h[1] = *(const h8*)(as16 + hl * CCH + kbase + 16 + half * 8);
        const _Float16* bp = WoT + (ncol0 + hl) * CCH + kbase + half * 8;
        b.h[0] = *(const h8*)(bp);
        b.h[1] = *(const h8*)(bp + 16);
        acc.v = __builtin_amdgcn_wmma_f32_16x16x32_f16(false, a.v, false, b.v,
                                                       (short)0, acc.v, false, false);
    }
    #pragma unroll
    for (int r = 0; r < 8; ++r)
        ytile[(half * 8 + r) * 132 + ncol0 + hl] = acc.e[r];
    __syncthreads();

    // LayerNorm: threads t = r*16 + c16 (same 16-lane half) cover row r
    const int r  = t >> 4;
    const int c0 = (t & 15) * 8;
    float s1 = 0.f, s2 = 0.f;
    #pragma unroll
    for (int j = 0; j < 8; ++j) {
        const float yv = ytile[r * 132 + c0 + j];
        s1 += yv; s2 += yv * yv;
    }
    #pragma unroll
    for (int off = 1; off < 16; off <<= 1) {
        s1 += __shfl_xor(s1, off, 32);
        s2 += __shfl_xor(s2, off, 32);
    }
    const float mu  = s1 * (1.f / 128.f);
    const float var = s2 * (1.f / 128.f) - mu * mu;
    const float rs  = rsqrtf(var + 1e-3f);
    #pragma unroll
    for (int j = 0; j < 8; ++j) {
        const int c = c0 + j;
        const float yv = ytile[r * 132 + c];
        out[(row0 + r) * CCH + c] = gamma[c] * (yv - mu) * rs + beta[c];
    }
}

extern "C" void kernel_launch(void* const* d_in, const int* in_sizes, int n_in,
                              void* d_out, int out_size, void* d_ws, size_t ws_size,
                              hipStream_t stream) {
    const float* x     = (const float*)d_in[0];
    const float* Wq    = (const float*)d_in[1];
    const float* Wk    = (const float*)d_in[2];
    const float* Wv    = (const float*)d_in[3];
    const float* Wo    = (const float*)d_in[4];
    const float* rel_h = (const float*)d_in[5];
    const float* rel_w = (const float*)d_in[6];
    const float* gamma = (const float*)d_in[7];
    const float* beta  = (const float*)d_in[8];

    char* ws = (char*)d_ws;
    float*    qkv   = (float*)ws;                                  // 24 MB
    _Float16* att16 = (_Float16*)(ws + (size_t)3 * NPIX * CCH * 4);// 4 MB
    _Float16* x16   = (_Float16*)(ws + (size_t)3 * NPIX * CCH * 4
                                     + (size_t)NPIX * CCH * 2);    // 4 MB
    _Float16* Wt    = (_Float16*)(ws + (size_t)3 * NPIX * CCH * 4
                                     + (size_t)2 * NPIX * CCH * 2);// 128 KB
    float* out = (float*)d_out;

    prep_weights<<<dim3(256), 256, 0, stream>>>(Wq, Wk, Wv, Wo, Wt);
    convert_x<<<dim3((NPIX * CCH) / 256), 256, 0, stream>>>(x, x16);
    qkv_proj_wmma<<<dim3(NPIX / 16, 3), 256, 0, stream>>>(x16, Wt, qkv);
    sasa_attn<<<dim3(NPIX), 128, 0, stream>>>(qkv,
                                              qkv + (size_t)1 * NPIX * CCH,
                                              qkv + (size_t)2 * NPIX * CCH,
                                              rel_h, rel_w, att16);
    out_proj_ln_wmma<<<dim3(NPIX / 16), 256, 0, stream>>>(att16, Wt + (size_t)3 * CCH * CCH,
                                                          gamma, beta, out);
}
